// AutoInt_21139829031304
// MI455X (gfx1250) — matrix operand order repaired and checked by reading, not compiled, not run on hardware
//
#include <hip/hip_runtime.h>

// AutoInt fused kernel for gfx1250 (MI455X): one wave32 per sample, all
// activations resident in wave-private LDS, all GEMMs via v_wmma_f32_16x16x32_f16.

typedef __attribute__((ext_vector_type(16))) _Float16 v16h;
typedef __attribute__((ext_vector_type(8)))  _Float16 v8h;
typedef __attribute__((ext_vector_type(8)))  float    v8f;

#define BATCH    16384
#define NUM_NUM  13
#define NUM_CAT  26
#define VOCAB    10000
#define E_DIM    64
#define LAYERS   3
#define HEADS    2
#define DHEAD    32
#define FIELDS   39
#define FPAD     48      // 3 tiles of 16 rows

// ---- WMMA wrapper: D = A(16x32 f16) * B(32x16 f16) + C(16x16 f32) ----
__device__ __forceinline__ v8f wmma_f16(v16h a, v16h b, v8f c) {
  return __builtin_amdgcn_wmma_f32_16x16x32_f16(
      /*neg_a=*/false, a, /*neg_b=*/false, b,
      /*c_mod=*/(short)0, c, /*reuse_a=*/false, /*reuse_b=*/false);
}

// A fragment (16x32, f16) from row-major buffer, tile origin (m0,k0), stride ld.
// ISA layout: lane<16 -> row m0+lane, halfs[0..7]=K k0..k0+7, halfs[8..15]=K k0+16..k0+23
//             lane>=16 -> same row, K+8 / K+24.
__device__ __forceinline__ v16h ldA(const _Float16* buf, int ld, int m0, int k0, int lane) {
  int r = lane & 15, hl = lane >> 4;
  const _Float16* p = buf + (m0 + r) * ld + k0 + hl * 8;
  v8h lo = *(const v8h*)p;
  v8h hi = *(const v8h*)(p + 16);
  v16h out;
#pragma unroll
  for (int i = 0; i < 8; ++i) { out[i] = lo[i]; out[8 + i] = hi[i]; }
  return out;
}

// B fragment (32x16, f16) where B[k][n] = Mat[n][k] (Mat row-major, stride ld).
// ISA layout: lane<16 -> col n0+lane, halfs[0..7]=K k0..k0+7, halfs[8..15]=K k0+16..k0+23;
// lane>=16 -> K+8 / K+24 (same order as ldA) -> two 16-byte loads.
__device__ __forceinline__ v16h ldBt(const _Float16* mat, int ld, int n0, int k0, int lane) {
  int c = lane & 15, hl = lane >> 4;
  const _Float16* p = mat + (n0 + c) * ld + k0 + hl * 8;
  v8h lo = *(const v8h*)p;
  v8h hi = *(const v8h*)(p + 16);
  v16h out;
#pragma unroll
  for (int i = 0; i < 8; ++i) { out[i] = lo[i]; out[8 + i] = hi[i]; }
  return out;
}

// Store 16x16 f32 D fragment as f16, normal row-major (stride ld).
__device__ __forceinline__ void stD16(_Float16* buf, int ld, int m0, int n0, v8f d, int lane) {
  int c = lane & 15, hl = lane >> 4;
#pragma unroll
  for (int v = 0; v < 8; ++v) buf[(m0 + v + 8 * hl) * ld + n0 + c] = (_Float16)d[v];
}

// Store transposed (buf[n][m]), stride ld.
__device__ __forceinline__ void stD16T(_Float16* buf, int ld, int m0, int n0, v8f d, int lane) {
  int c = lane & 15, hl = lane >> 4;
#pragma unroll
  for (int v = 0; v < 8; ++v) buf[(n0 + c) * ld + m0 + v + 8 * hl] = (_Float16)d[v];
}

// Store scaled f32 D fragment (scores), stride ld.
__device__ __forceinline__ void stDf32(float* buf, int ld, int m0, int n0, v8f d, float s, int lane) {
  int c = lane & 15, hl = lane >> 4;
#pragma unroll
  for (int v = 0; v < 8; ++v) buf[(m0 + v + 8 * hl) * ld + n0 + c] = d[v] * s;
}

// ---- weight pre-conversion f32 -> f16 into workspace ----
// ws layout: for l in 0..2: [Wq, Wk, Wv, Wo, Wres], each 64x64 (out,in) row-major.
__global__ void convert_weights(const float* wq, const float* wk, const float* wv,
                                const float* wo, const float* wr, _Float16* w16) {
  int i = blockIdx.x * blockDim.x + threadIdx.x;
  if (i >= LAYERS * 5 * E_DIM * E_DIM) return;
  int l = i / (5 * 4096);
  int rem = i % (5 * 4096);
  int m = rem / 4096;
  int j = rem % 4096;
  float v;
  switch (m) {
    case 0:  v = wq[l * 4096 + j]; break;
    case 1:  v = wk[l * 4096 + j]; break;
    case 2:  v = wv[l * 4096 + j]; break;
    case 3:  v = wo[l * 4096 + j]; break;
    default: v = wr[l * 4096 + j]; break;
  }
  w16[i] = (_Float16)v;
}

// ---- fused AutoInt: one wave32 per sample ----
__global__ __launch_bounds__(32)
void autoint_fused(const float* __restrict__ num_f, const int* __restrict__ cat_f,
                   const float* __restrict__ W_num, const float* __restrict__ b_num,
                   const float* __restrict__ cat_tab, const _Float16* __restrict__ w16,
                   const float* __restrict__ W_final, const float* __restrict__ b_final,
                   float* __restrict__ out) {
  __shared__ __align__(16) _Float16 X[FPAD * E_DIM];     // activations (48x64)
  __shared__ __align__(16) _Float16 Qb[FPAD * E_DIM];    // Q
  __shared__ __align__(16) _Float16 Kb[FPAD * E_DIM];    // K
  __shared__ __align__(16) _Float16 Vt[E_DIM * E_DIM];   // V transposed (d x kf), kf padded to 64
  __shared__ __align__(16) _Float16 Ob[FPAD * E_DIM];    // attention output
  __shared__ __align__(16) _Float16 Pb[FPAD * E_DIM];    // softmax probs / X_new temp
  __shared__ __align__(16) float    Sb[FPAD * FPAD];     // raw scores (f32)

  const int lane = threadIdx.x;
  const int s = blockIdx.x;

  // ---- build embeddings into X (f16), zero padding rows ----
  for (int i = lane; i < NUM_NUM * E_DIM; i += 32) {
    int f = i >> 6;
    X[i] = (_Float16)(num_f[s * NUM_NUM + f] * W_num[i] + b_num[i]);
  }
  for (int i = lane; i < NUM_CAT * E_DIM; i += 32) {
    int f = i >> 6, e = i & 63;
    int idx = cat_f[s * NUM_CAT + f];
    X[(NUM_NUM + f) * E_DIM + e] =
        (_Float16)cat_tab[((size_t)f * VOCAB + idx) * E_DIM + e];
  }
  for (int i = lane; i < (FPAD - FIELDS) * E_DIM; i += 32)
    X[FIELDS * E_DIM + i] = (_Float16)0.f;
  // zero Vt pad columns (kf = 48..63) once; never rewritten.
  for (int i = lane; i < E_DIM * 16; i += 32) {
    int d = i >> 4, kf = FPAD + (i & 15);
    Vt[d * E_DIM + kf] = (_Float16)0.f;
  }
  __syncthreads();

  const float inv_sqrt_d = 0.17677669529663687f;  // 1/sqrt(32)

  for (int l = 0; l < LAYERS; ++l) {
    const _Float16* Wq = w16 + (size_t)(l * 5 + 0) * 4096;
    const _Float16* Wk = w16 + (size_t)(l * 5 + 1) * 4096;
    const _Float16* Wv = w16 + (size_t)(l * 5 + 2) * 4096;
    const _Float16* Wo = w16 + (size_t)(l * 5 + 3) * 4096;
    const _Float16* Wr = w16 + (size_t)(l * 5 + 4) * 4096;

    // ---- Q = X Wq^T, K = X Wk^T, V^T = (X Wv^T)^T ----
    for (int m0 = 0; m0 < FPAD; m0 += 16) {
      v16h a0 = ldA(X, E_DIM, m0, 0, lane);
      v16h a1 = ldA(X, E_DIM, m0, 32, lane);
#pragma unroll
      for (int n0 = 0; n0 < E_DIM; n0 += 16) {
        v8f acc = {};
        acc = wmma_f16(a0, ldBt(Wq, E_DIM, n0, 0, lane), acc);
        acc = wmma_f16(a1, ldBt(Wq, E_DIM, n0, 32, lane), acc);
        stD16(Qb, E_DIM, m0, n0, acc, lane);
        v8f acck = {};
        acck = wmma_f16(a0, ldBt(Wk, E_DIM, n0, 0, lane), acck);
        acck = wmma_f16(a1, ldBt(Wk, E_DIM, n0, 32, lane), acck);
        stD16(Kb, E_DIM, m0, n0, acck, lane);
        v8f accv = {};
        accv = wmma_f16(a0, ldBt(Wv, E_DIM, n0, 0, lane), accv);
        accv = wmma_f16(a1, ldBt(Wv, E_DIM, n0, 32, lane), accv);
        stD16T(Vt, E_DIM, m0, n0, accv, lane);  // Vt[d][kf]
      }
    }
    __syncthreads();

    for (int h = 0; h < HEADS; ++h) {
      const int d0 = h * DHEAD;
      // ---- scores = (Q_h K_h^T) / sqrt(D)  (48x48, f32) ----
      for (int m0 = 0; m0 < FPAD; m0 += 16) {
        v16h aq = ldA(Qb, E_DIM, m0, d0, lane);
#pragma unroll
        for (int n0 = 0; n0 < FPAD; n0 += 16) {
          v8f acc = {};
          acc = wmma_f16(aq, ldBt(Kb, E_DIM, n0, d0, lane), acc);
          stDf32(Sb, FPAD, m0, n0, acc, inv_sqrt_d, lane);
        }
      }
      __syncthreads();
      // ---- masked softmax over kf in [0,39); P zero-padded to 48x64 ----
      for (int r = lane; r < FPAD; r += 32) {
        if (r < FIELDS) {
          float mx = -1e30f;
          for (int c = 0; c < FIELDS; ++c) mx = fmaxf(mx, Sb[r * FPAD + c]);
          float sum = 0.f;
          for (int c = 0; c < FIELDS; ++c) {
            float t = __expf(Sb[r * FPAD + c] - mx);
            Sb[r * FPAD + c] = t;
            sum += t;
          }
          float inv = 1.f / sum;
          for (int c = 0; c < FIELDS; ++c) Pb[r * E_DIM + c] = (_Float16)(Sb[r * FPAD + c] * inv);
          for (int c = FIELDS; c < E_DIM; ++c) Pb[r * E_DIM + c] = (_Float16)0.f;
        } else {
          for (int c = 0; c < E_DIM; ++c) Pb[r * E_DIM + c] = (_Float16)0.f;
        }
      }
      __syncthreads();
      // ---- OUT[:, d0:d0+32] = P V_h  (K padded to 64) ----
      for (int m0 = 0; m0 < FPAD; m0 += 16) {
        v16h p0 = ldA(Pb, E_DIM, m0, 0, lane);
        v16h p1 = ldA(Pb, E_DIM, m0, 32, lane);
#pragma unroll
        for (int n0 = 0; n0 < DHEAD; n0 += 16) {
          v8f acc = {};
          acc = wmma_f16(p0, ldBt(Vt, E_DIM, d0 + n0, 0, lane), acc);
          acc = wmma_f16(p1, ldBt(Vt, E_DIM, d0 + n0, 32, lane), acc);
          stD16(Ob, E_DIM, m0, d0 + n0, acc, lane);
        }
      }
      __syncthreads();
    }

    // ---- X_new = OUT Wo^T + X Wres^T  (into Pb temp, then copy back) ----
    for (int m0 = 0; m0 < FPAD; m0 += 16) {
      v16h o0 = ldA(Ob, E_DIM, m0, 0, lane);
      v16h o1 = ldA(Ob, E_DIM, m0, 32, lane);
      v16h x0 = ldA(X, E_DIM, m0, 0, lane);
      v16h x1 = ldA(X, E_DIM, m0, 32, lane);
#pragma unroll
      for (int n0 = 0; n0 < E_DIM; n0 += 16) {
        v8f acc = {};
        acc = wmma_f16(o0, ldBt(Wo, E_DIM, n0, 0, lane), acc);
        acc = wmma_f16(o1, ldBt(Wo, E_DIM, n0, 32, lane), acc);
        acc = wmma_f16(x0, ldBt(Wr, E_DIM, n0, 0, lane), acc);
        acc = wmma_f16(x1, ldBt(Wr, E_DIM, n0, 32, lane), acc);
        stD16(Pb, E_DIM, m0, n0, acc, lane);
      }
    }
    __syncthreads();
    for (int i = lane; i < FPAD * E_DIM / 8; i += 32)
      *((v8h*)X + i) = *((const v8h*)Pb + i);
    __syncthreads();
  }

  // ---- readout: sigmoid(flat(X[0:39]) . W_final + b) ----
  float acc = 0.f;
  for (int i = lane; i < FIELDS * E_DIM; i += 32)
    acc += (float)X[i] * W_final[i];
#pragma unroll
  for (int off = 16; off > 0; off >>= 1) acc += __shfl_down(acc, off, 32);
  if (lane == 0) {
    float logit = acc + b_final[0];
    out[s] = 1.f / (1.f + __expf(-logit));
  }
}

extern "C" void kernel_launch(void* const* d_in, const int* in_sizes, int n_in,
                              void* d_out, int out_size, void* d_ws, size_t ws_size,
                              hipStream_t stream) {
  (void)in_sizes; (void)n_in; (void)out_size; (void)ws_size;
  const float* num_f   = (const float*)d_in[0];
  const int*   cat_f   = (const int*)d_in[1];
  const float* W_num   = (const float*)d_in[2];
  const float* b_num   = (const float*)d_in[3];
  const float* cat_tab = (const float*)d_in[4];
  const float* WQ      = (const float*)d_in[5];
  const float* WK      = (const float*)d_in[6];
  const float* WV      = (const float*)d_in[7];
  const float* WO      = (const float*)d_in[8];
  const float* WR      = (const float*)d_in[9];
  const float* W_final = (const float*)d_in[10];
  const float* b_final = (const float*)d_in[11];
  _Float16* w16 = (_Float16*)d_ws;   // 3*5*4096 halfs = 122880 bytes

  const int nw = LAYERS * 5 * E_DIM * E_DIM;
  convert_weights<<<(nw + 255) / 256, 256, 0, stream>>>(WQ, WK, WV, WO, WR, w16);
  autoint_fused<<<BATCH, 32, 0, stream>>>(num_f, cat_f, W_num, b_num, cat_tab,
                                          w16, W_final, b_final, (float*)d_out);
}
